// FixLocator_88304527606637
// MI455X (gfx1250) — compile-verified, matrix operand order checked by reading, not run on hardware
//
#include <hip/hip_runtime.h>
#include <hip/hip_bf16.h>

// ---------------------------------------------------------------------------
// FixLocator for MI455X (gfx1250): bf16 WMMA GEMM core + recurrent gate math
// ---------------------------------------------------------------------------

typedef __bf16 bf16_t;
typedef __attribute__((ext_vector_type(16))) __bf16 v16bf;
typedef __attribute__((ext_vector_type(8)))  __bf16 v8bf;
typedef __attribute__((ext_vector_type(8)))  float  v8f;
typedef __attribute__((ext_vector_type(4)))  float  v4f;

#define NM_N    4000
#define NS_N    10000
#define T_LEN   16
#define F_DIM   256
#define H_DIM   256
#define M_DIM   768
#define M3_DIM  2304
#define CC_DIM  512
#define EM_CNT  128000
#define ES_CNT  320000

// ---------------------------- utility kernels ------------------------------

__global__ void k_fill(float* __restrict__ p, float v, long long n) {
  long long i = (long long)blockIdx.x * blockDim.x + threadIdx.x;
  if (i < n) p[i] = v;
}

__global__ void k_cvt_bf16(bf16_t* __restrict__ dst, const float* __restrict__ src,
                           long long n) {
  long long i = (long long)blockIdx.x * blockDim.x + threadIdx.x;
  if (i < n) dst[i] = (bf16_t)src[i];
}

// ------------------------------ WMMA GEMM ----------------------------------
// C[row,col] = A[N,K](f32, row stride a_rstride) x W[O,K](bf16)^T + bias
// optionally accumulating into existing C.  Grid: (N/16, O/256), block 128.
// Per wave: 16x64 output tile, K in steps of 32 via v_wmma_f32_16x16x32_bf16.
// A-fragment layout per CDNA5 ISA (16-bit A 16x32):
//   lanes 0..15  : row M=lane,   K = {k..k+7,  k+16..k+23}
//   lanes 16..31 : row M=lane-16,K = {k+8..k+15,k+24..k+31}
// B passed as the [O x K] mirror of A (weight rows are output columns).

__global__ __launch_bounds__(128)
void k_gemm_wmma(const float* __restrict__ A, long long a_rstride,
                 const bf16_t* __restrict__ W,
                 const float* __restrict__ bias,
                 float* __restrict__ C, long long c_rstride,
                 int K, int accumulate)
{
  const int lane = threadIdx.x & 31;
  const int wave = threadIdx.x >> 5;
  const int lrow = lane & 15;
  const int hi   = (lane >> 4) & 1;

  const long long row0 = (long long)blockIdx.x * 16;
  const long long col0 = (long long)blockIdx.y * 256 + wave * 64;

  const float* __restrict__ arow = A + (row0 + lrow) * a_rstride;
  const bf16_t* __restrict__ wr0 = W + (size_t)(col0 +  0 + lrow) * (size_t)K;
  const bf16_t* __restrict__ wr1 = W + (size_t)(col0 + 16 + lrow) * (size_t)K;
  const bf16_t* __restrict__ wr2 = W + (size_t)(col0 + 32 + lrow) * (size_t)K;
  const bf16_t* __restrict__ wr3 = W + (size_t)(col0 + 48 + lrow) * (size_t)K;

  v8f acc0 = {0.f,0.f,0.f,0.f,0.f,0.f,0.f,0.f};
  v8f acc1 = acc0, acc2 = acc0, acc3 = acc0;

  for (int k = 0; k < K; k += 32) {
    const int ka = k + (hi ? 8 : 0);

    // ---- A fragment: 16 f32 -> bf16 in-register -------------------------
    v4f a0 = *(const v4f*)(arow + ka);
    v4f a1 = *(const v4f*)(arow + ka + 4);
    v4f a2 = *(const v4f*)(arow + ka + 16);
    v4f a3 = *(const v4f*)(arow + ka + 20);
    v16bf af;
#pragma unroll
    for (int i = 0; i < 4; ++i) {
      af[i +  0] = (bf16_t)a0[i];
      af[i +  4] = (bf16_t)a1[i];
      af[i +  8] = (bf16_t)a2[i];
      af[i + 12] = (bf16_t)a3[i];
    }

    // ---- B fragments: 4 tiles of 16 cols, bf16 direct -------------------
    v8bf lo0 = *(const v8bf*)(wr0 + ka);  v8bf hi0 = *(const v8bf*)(wr0 + ka + 16);
    v8bf lo1 = *(const v8bf*)(wr1 + ka);  v8bf hi1 = *(const v8bf*)(wr1 + ka + 16);
    v8bf lo2 = *(const v8bf*)(wr2 + ka);  v8bf hi2 = *(const v8bf*)(wr2 + ka + 16);
    v8bf lo3 = *(const v8bf*)(wr3 + ka);  v8bf hi3 = *(const v8bf*)(wr3 + ka + 16);
    v16bf b0, b1, b2, b3;
#pragma unroll
    for (int i = 0; i < 8; ++i) {
      b0[i] = lo0[i]; b0[i + 8] = hi0[i];
      b1[i] = lo1[i]; b1[i + 8] = hi1[i];
      b2[i] = lo2[i]; b2[i + 8] = hi2[i];
      b3[i] = lo3[i]; b3[i + 8] = hi3[i];
    }

    acc0 = __builtin_amdgcn_wmma_f32_16x16x32_bf16(false, af, false, b0, (short)0, acc0, false, false);
    acc1 = __builtin_amdgcn_wmma_f32_16x16x32_bf16(false, af, false, b1, (short)0, acc1, false, false);
    acc2 = __builtin_amdgcn_wmma_f32_16x16x32_bf16(false, af, false, b2, (short)0, acc2, false, false);
    acc3 = __builtin_amdgcn_wmma_f32_16x16x32_bf16(false, af, false, b3, (short)0, acc3, false, false);
  }

  // ---- store: C layout VGPR r -> row = row0 + r + hi*8, col = col0+j*16+lrow
  v8f accs[4] = {acc0, acc1, acc2, acc3};
#pragma unroll
  for (int j = 0; j < 4; ++j) {
    const long long col = col0 + j * 16 + lrow;
    const float badd = bias ? bias[col] : 0.0f;
#pragma unroll
    for (int r = 0; r < 8; ++r) {
      const long long row = row0 + r + (hi ? 8 : 0);
      const long long idx = row * c_rstride + col;
      float v = accs[j][r] + badd;
      if (accumulate) v += C[idx];
      C[idx] = v;
    }
  }
}

// ------------------------- recurrent gate kernels --------------------------

__device__ __forceinline__ float sigf(float x) { return 1.0f / (1.0f + __expf(-x)); }

__global__ void k_gru_step(const float* __restrict__ gi, const float* __restrict__ gh,
                           float* __restrict__ h, int N) {
  long long i = (long long)blockIdx.x * blockDim.x + threadIdx.x;
  if (i >= (long long)N * M_DIM) return;
  const int n = (int)(i / M_DIM);
  const int j = (int)(i % M_DIM);
  const float* gir = gi + (long long)n * M3_DIM;
  const float* ghr = gh + (long long)n * M3_DIM;
  const float r  = sigf(gir[j]            + ghr[j]);
  const float z  = sigf(gir[M_DIM + j]    + ghr[M_DIM + j]);
  const float nn = tanhf(gir[2*M_DIM + j] + r * ghr[2*M_DIM + j]);
  const float hv = h[i];
  h[i] = (1.0f - z) * nn + z * hv;
}

__global__ void k_tree_step(const float* __restrict__ iou, const float* __restrict__ fb,
                            float* __restrict__ c, float* __restrict__ h, int N) {
  long long idx = (long long)blockIdx.x * blockDim.x + threadIdx.x;
  if (idx >= (long long)N * M_DIM) return;
  const int n = (int)(idx / M_DIM);
  const int j = (int)(idx % M_DIM);
  const float* ir = iou + (long long)n * M3_DIM;
  const float i = sigf(ir[j]);
  const float o = sigf(ir[M_DIM + j]);
  const float u = tanhf(ir[2*M_DIM + j]);
  const float f = sigf(fb[idx]);
  const float cv = i * u + f * c[idx];
  c[idx] = cv;
  h[idx] = o * tanhf(cv);
}

// ------------------------------- GCN kernels -------------------------------

__global__ void k_deg(const int* __restrict__ dst, float* __restrict__ deg, int E) {
  int e = blockIdx.x * blockDim.x + threadIdx.x;
  if (e < E) atomicAdd(&deg[dst[e]], 1.0f);
}

__global__ void k_rsqrt_inplace(float* __restrict__ deg, int N) {
  int n = blockIdx.x * blockDim.x + threadIdx.x;
  if (n < N) deg[n] = rsqrtf(deg[n]);   // deg >= 1 always (self loop)
}

__global__ void k_scatter(const int* __restrict__ src, const int* __restrict__ dst,
                          const float* __restrict__ xw, const float* __restrict__ dinv,
                          float* __restrict__ out, int E) {
  const int e = blockIdx.x;
  const int cidx = blockIdx.y * blockDim.x + threadIdx.x;
  if (e >= E || cidx >= M_DIM) return;
  const int s = src[e];
  const int d = dst[e];
  const float norm = dinv[s] * dinv[d];
  atomicAdd(&out[(long long)d * M_DIM + cidx], xw[(long long)s * M_DIM + cidx] * norm);
}

__global__ void k_gcn_final(const float* __restrict__ tmp, const float* __restrict__ xw,
                            const float* __restrict__ dinv, const float* __restrict__ b,
                            float* __restrict__ out, int N, int relu) {
  long long idx = (long long)blockIdx.x * blockDim.x + threadIdx.x;
  if (idx >= (long long)N * M_DIM) return;
  const int n = (int)(idx / M_DIM);
  const int j = (int)(idx % M_DIM);
  const float di = dinv[n];
  float v = tmp[idx] + xw[idx] * di * di + b[j];   // self-loop term + bias
  if (relu) v = fmaxf(v, 0.0f);
  out[idx] = v;
}

// ------------------------------- head --------------------------------------

__global__ void k_head(const float* __restrict__ x, const float* __restrict__ w,
                       const float* __restrict__ b, float* __restrict__ out, int N) {
  int n = blockIdx.x * blockDim.x + threadIdx.x;
  if (n >= N) return;
  const float* xr = x + (long long)n * M_DIM;
  float s0 = b[0], s1 = b[1];
  for (int j = 0; j < M_DIM; ++j) {
    const float xv = xr[j];
    s0 += xv * w[j];
    s1 += xv * w[M_DIM + j];
  }
  const float m  = fmaxf(s0, s1);
  const float e0 = __expf(s0 - m);
  const float e1 = __expf(s1 - m);
  const float inv = 1.0f / (e0 + e1);
  out[2 * n]     = e0 * inv;
  out[2 * n + 1] = e1 * inv;
}

// ---------------------------------------------------------------------------
// host orchestration
// ---------------------------------------------------------------------------

// flattened-input indices (setup_inputs() insertion order, recursive)
enum {
  IN_F1 = 0, IN_F2, IN_F3, IN_F4, IN_F5, IN_F6, IN_EIM, IN_EIS,
  P_GRU1_WIH = 8,  P_GRU1_WHH, P_GRU1_BIH, P_GRU1_BHH,
  P_GRU2_WIH = 12, P_GRU2_WHH, P_GRU2_BIH, P_GRU2_BHH,
  P_TREE1 = 16,   // ioux.w, ioux.b, iouh.w, iouh.b, fx.w, fx.b, fh.w, fh.b
  P_TREE2 = 24,
  P_TREE3 = 32,
  P_RS1 = 40, P_RS2 = 42, P_RS3 = 44, P_RS4 = 46,
  P_RS5 = 48, P_RS6 = 50, P_RS7 = 52, P_RS8 = 54,
  P_GCNM = 56,    // 3 x (w, b)
  P_GCNS = 62
};

extern "C" void kernel_launch(void* const* d_in, const int* in_sizes, int n_in,
                              void* d_out, int out_size, void* d_ws, size_t ws_size,
                              hipStream_t stream) {
  (void)in_sizes; (void)n_in; (void)out_size; (void)ws_size;
  char* ws = (char*)d_ws;
  size_t off = 0;
  auto alloc = [&](size_t bytes) -> void* {
    off = (off + 255) & ~(size_t)255;
    void* p = ws + off;
    off += bytes;
    return p;
  };

  auto fin = [&](int idx) -> const float* { return (const float*)d_in[idx]; };

  // ---- convert all GEMM weights to bf16 once per launch -------------------
  auto cvtW = [&](int idx, long long nelem) -> bf16_t* {
    bf16_t* p = (bf16_t*)alloc((size_t)nelem * sizeof(bf16_t));
    long long g = (nelem + 255) / 256;
    k_cvt_bf16<<<dim3((unsigned)g), 256, 0, stream>>>(p, fin(idx), nelem);
    return p;
  };

  bf16_t* gru1_wih = cvtW(P_GRU1_WIH, (long long)M3_DIM * F_DIM);
  bf16_t* gru1_whh = cvtW(P_GRU1_WHH, (long long)M3_DIM * M_DIM);
  bf16_t* gru2_wih = cvtW(P_GRU2_WIH, (long long)M3_DIM * F_DIM);
  bf16_t* gru2_whh = cvtW(P_GRU2_WHH, (long long)M3_DIM * M_DIM);

  bf16_t *tr_ioux[3], *tr_iouh[3], *tr_fx[3], *tr_fh[3];
  const int tree_base[3] = {P_TREE1, P_TREE2, P_TREE3};
  for (int i = 0; i < 3; ++i) {
    tr_ioux[i] = cvtW(tree_base[i] + 0, (long long)M3_DIM * F_DIM);
    tr_iouh[i] = cvtW(tree_base[i] + 2, (long long)M3_DIM * M_DIM);
    tr_fx[i]   = cvtW(tree_base[i] + 4, (long long)M_DIM * F_DIM);
    tr_fh[i]   = cvtW(tree_base[i] + 6, (long long)M_DIM * M_DIM);
  }

  bf16_t* rs_w[6];
  const int rs_idx[6] = {P_RS1, P_RS2, P_RS3, P_RS4, P_RS5, P_RS6};
  const long long rs_k[6] = {M_DIM, M_DIM, M_DIM, CC_DIM, M_DIM, M_DIM};
  for (int i = 0; i < 6; ++i) rs_w[i] = cvtW(rs_idx[i], (long long)H_DIM * rs_k[i]);

  bf16_t *gcnm_w[3], *gcns_w[3];
  for (int i = 0; i < 3; ++i) {
    gcnm_w[i] = cvtW(P_GCNM + 2 * i, (long long)M_DIM * M_DIM);
    gcns_w[i] = cvtW(P_GCNS + 2 * i, (long long)M_DIM * M_DIM);
  }

  // ---- workspace buffers --------------------------------------------------
  float* gbuf1 = (float*)alloc((size_t)NS_N * M3_DIM * 4);  // gi / iou
  float* gbuf2 = (float*)alloc((size_t)NS_N * M3_DIM * 4);  // gh
  float* fbuf  = (float*)alloc((size_t)NS_N * M_DIM * 4);   // tree f-gate
  float* hbuf  = (float*)alloc((size_t)NS_N * M_DIM * 4);
  float* cbuf  = (float*)alloc((size_t)NS_N * M_DIM * 4);
  float* mf    = (float*)alloc((size_t)NM_N * M_DIM * 4);
  float* mnext = (float*)alloc((size_t)NM_N * M_DIM * 4);
  float* mtmp  = (float*)alloc((size_t)NM_N * M_DIM * 4);
  float* xwm   = (float*)alloc((size_t)NM_N * M_DIM * 4);
  float* sf    = (float*)alloc((size_t)NS_N * M_DIM * 4);
  float* snext = (float*)alloc((size_t)NS_N * M_DIM * 4);
  float* stmp  = (float*)alloc((size_t)NS_N * M_DIM * 4);
  float* xws   = (float*)alloc((size_t)NS_N * M_DIM * 4);
  float* degm  = (float*)alloc((size_t)NM_N * 4);
  float* degs  = (float*)alloc((size_t)NS_N * 4);

  auto fill = [&](float* p, float v, long long n) {
    k_fill<<<dim3((unsigned)((n + 255) / 256)), 256, 0, stream>>>(p, v, n);
  };

  auto gemm = [&](const float* A, long long arst, const bf16_t* W, const float* bias,
                  float* C, long long crst, int N, int K, int O, int accflag) {
    dim3 grid((unsigned)(N / 16), (unsigned)(O / 256));
    k_gemm_wmma<<<grid, 128, 0, stream>>>(A, arst, W, bias, C, crst, K, accflag);
  };

  // ---- GRU ----------------------------------------------------------------
  auto run_gru = [&](const float* X, int N, const bf16_t* wih, const bf16_t* whh,
                     const float* bih, const float* bhh) {
    const long long hn = (long long)N * M_DIM;
    fill(hbuf, 0.0f, hn);
    for (int t = 0; t < T_LEN; ++t) {
      gemm(X + (long long)t * F_DIM, (long long)T_LEN * F_DIM, wih, bih,
           gbuf1, M3_DIM, N, F_DIM, M3_DIM, 0);
      gemm(hbuf, M_DIM, whh, bhh, gbuf2, M3_DIM, N, M_DIM, M3_DIM, 0);
      k_gru_step<<<dim3((unsigned)((hn + 255) / 256)), 256, 0, stream>>>(gbuf1, gbuf2, hbuf, N);
    }
  };

  // ---- chain Tree-LSTM ----------------------------------------------------
  auto run_tree = [&](const float* X, int N, int ti) {
    const long long hn = (long long)N * M_DIM;
    fill(hbuf, 0.0f, hn);
    fill(cbuf, 0.0f, hn);
    const float* bioux = fin(tree_base[ti] + 1);
    const float* biouh = fin(tree_base[ti] + 3);
    const float* bfx   = fin(tree_base[ti] + 5);
    const float* bfh   = fin(tree_base[ti] + 7);
    for (int t = 0; t < T_LEN; ++t) {
      const float* Xt = X + (long long)t * F_DIM;
      gemm(Xt,   (long long)T_LEN * F_DIM, tr_ioux[ti], bioux, gbuf1, M3_DIM, N, F_DIM, M3_DIM, 0);
      gemm(hbuf, M_DIM,                    tr_iouh[ti], biouh, gbuf1, M3_DIM, N, M_DIM, M3_DIM, 1);
      gemm(Xt,   (long long)T_LEN * F_DIM, tr_fx[ti],   bfx,   fbuf,  M_DIM,  N, F_DIM, M_DIM,  0);
      gemm(hbuf, M_DIM,                    tr_fh[ti],   bfh,   fbuf,  M_DIM,  N, M_DIM, M_DIM,  1);
      k_tree_step<<<dim3((unsigned)((hn + 255) / 256)), 256, 0, stream>>>(gbuf1, fbuf, cbuf, hbuf, N);
    }
  };

  // ---- feature branches (write into column slices of m_f / s_f) ----------
  run_gru(fin(IN_F1), NM_N, gru1_wih, gru1_whh, fin(P_GRU1_BIH), fin(P_GRU1_BHH));
  gemm(hbuf, M_DIM, rs_w[0], fin(P_RS1 + 1), mf + 0,   M_DIM, NM_N, M_DIM, H_DIM, 0);

  run_tree(fin(IN_F2), NM_N, 0);
  gemm(hbuf, M_DIM, rs_w[1], fin(P_RS2 + 1), mf + 256, M_DIM, NM_N, M_DIM, H_DIM, 0);

  run_tree(fin(IN_F3), NM_N, 1);
  gemm(hbuf, M_DIM, rs_w[2], fin(P_RS3 + 1), mf + 512, M_DIM, NM_N, M_DIM, H_DIM, 0);

  gemm(fin(IN_F4), CC_DIM, rs_w[3], fin(P_RS4 + 1), sf + 0, M_DIM, NS_N, CC_DIM, H_DIM, 0);

  run_tree(fin(IN_F5), NS_N, 2);
  gemm(hbuf, M_DIM, rs_w[4], fin(P_RS5 + 1), sf + 256, M_DIM, NS_N, M_DIM, H_DIM, 0);

  run_gru(fin(IN_F6), NS_N, gru2_wih, gru2_whh, fin(P_GRU2_BIH), fin(P_GRU2_BHH));
  gemm(hbuf, M_DIM, rs_w[5], fin(P_RS6 + 1), sf + 512, M_DIM, NS_N, M_DIM, H_DIM, 0);

  // ---- GCN stacks ---------------------------------------------------------
  auto run_gcn = [&](float* xcur, float* xnext, float* xw, float* tmp, float* deg,
                     const int* ei, int E, int N,
                     const bf16_t* w, const float* b, int relu) {
    const long long nm = (long long)N * M_DIM;
    gemm(xcur, M_DIM, w, nullptr, xw, M_DIM, N, M_DIM, M_DIM, 0);
    fill(deg, 1.0f, N);   // self loop
    k_deg<<<dim3((unsigned)((E + 255) / 256)), 256, 0, stream>>>(ei + E, deg, E);
    k_rsqrt_inplace<<<dim3((unsigned)((N + 255) / 256)), 256, 0, stream>>>(deg, N);
    fill(tmp, 0.0f, nm);
    dim3 sg((unsigned)E, (unsigned)(M_DIM / 256));
    k_scatter<<<sg, 256, 0, stream>>>(ei, ei + E, xw, deg, tmp, E);
    k_gcn_final<<<dim3((unsigned)((nm + 255) / 256)), 256, 0, stream>>>(tmp, xw, deg, b, xnext, N, relu);
  };

  const int* eim = (const int*)d_in[IN_EIM];
  const int* eis = (const int*)d_in[IN_EIS];

  // layer 0: mf -> mnext, layer 1: mnext -> mf, layer 2: mf -> mnext
  run_gcn(mf,    mnext, xwm, mtmp, degm, eim, EM_CNT, NM_N, gcnm_w[0], fin(P_GCNM + 1), 1);
  run_gcn(mnext, mf,    xwm, mtmp, degm, eim, EM_CNT, NM_N, gcnm_w[1], fin(P_GCNM + 3), 1);
  run_gcn(mf,    mnext, xwm, mtmp, degm, eim, EM_CNT, NM_N, gcnm_w[2], fin(P_GCNM + 5), 0);

  run_gcn(sf,    snext, xws, stmp, degs, eis, ES_CNT, NS_N, gcns_w[0], fin(P_GCNS + 1), 1);
  run_gcn(snext, sf,    xws, stmp, degs, eis, ES_CNT, NS_N, gcns_w[1], fin(P_GCNS + 3), 1);
  run_gcn(sf,    snext, xws, stmp, degs, eis, ES_CNT, NS_N, gcns_w[2], fin(P_GCNS + 5), 0);

  // ---- heads: resize7/8 + softmax ----------------------------------------
  float* out = (float*)d_out;
  k_head<<<dim3((unsigned)((NM_N + 255) / 256)), 256, 0, stream>>>(
      mnext, fin(P_RS7), fin(P_RS7 + 1), out, NM_N);
  k_head<<<dim3((unsigned)((NS_N + 255) / 256)), 256, 0, stream>>>(
      snext, fin(P_RS8), fin(P_RS8 + 1), out + 2 * NM_N, NS_N);
}
